// NonparametricPrototypes_87497073754720
// MI455X (gfx1250) — matrix operand order repaired and checked by reading, not compiled, not run on hardware
//
#include <hip/hip_runtime.h>
#include <hip/hip_bf16.h>
#include <math.h>

typedef __attribute__((ext_vector_type(16))) __bf16 v16bf;
typedef __attribute__((ext_vector_type(8)))  float  v8f;
typedef __attribute__((ext_vector_type(4)))  unsigned int v4u;
typedef __attribute__((ext_vector_type(4)))  unsigned int u32x4;
typedef __attribute__((ext_vector_type(8)))  int i32x8;
typedef __attribute__((ext_vector_type(4)))  int i32x4;

union Frag { v16bf v; v4u u[2]; };

#define BN_   (16 * 16384)   // B*N rows
#define C_    256
#define K_    256
#define ALPHA 0.1f
#define MOM   0.999f

// ---- compile-time probes (reported via stderr on successful compile) ----
#if __has_builtin(__builtin_amdgcn_tensor_load_to_lds)
#warning "PROBE: tensor_load_to_lds builtin PRESENT -> TDM path compiled"
#if __has_include(<hip/amd_detail/amd_gfx1250_TDM.h>)
#warning "PROBE: TDM header present -> using 6-arg builtin form"
#else
#warning "PROBE: TDM header absent -> using 5-arg builtin form"
#endif
#else
#warning "PROBE: tensor_load_to_lds builtin ABSENT -> fallback copy path"
#endif
#if __has_builtin(__builtin_amdgcn_s_wait_tensorcnt)
#warning "PROBE: s_wait_tensorcnt builtin PRESENT"
#else
#warning "PROBE: s_wait_tensorcnt builtin ABSENT"
#endif

// -------- workspace layout --------
// [0,        131072)  : p_bf16  (K*C bf16)
// [131072,   393216)  : sums    (K*C f32)
// [393216,   394240)  : counts  (K  f32)

// -------- dynamic LDS layout (assign_kernel) --------
// [0,      65536)  : x_raw  (64 rows x 256 f32)  <- TDM destination
// [65536,  98304)  : xbf    (64 rows x 256 bf16, normalized)
// [98304,  98560)  : inv_s  (64 f32)
// [98560,  98816)  : lab_s  (64 i32)
#define SMEM_BYTES 98816

// Kernel 1: normalize prototypes -> bf16; zero sums/counts.
__global__ void __launch_bounds__(256)
proto_prep(const float* __restrict__ protos,
           __bf16* __restrict__ pbf,
           float* __restrict__ sums,
           float* __restrict__ counts) {
    const int k = blockIdx.x;
    const int c = threadIdx.x;
    const float v = protos[k * C_ + c];
    float ss = v * v;
#pragma unroll
    for (int off = 16; off >= 1; off >>= 1) ss += __shfl_xor(ss, off, 32);
    __shared__ float part[8];
    const int w = c >> 5, ln = c & 31;
    if (ln == 0) part[w] = ss;
    __syncthreads();
    float tot = 0.f;
#pragma unroll
    for (int i = 0; i < 8; ++i) tot += part[i];
    const float inv = 1.0f / fmaxf(sqrtf(tot), 1e-12f);
    pbf[k * C_ + c]  = (__bf16)(v * inv);
    sums[k * C_ + c] = 0.0f;
    if (c == 0) counts[k] = 0.0f;
}

// Kernel 2: TDM-stage x tile -> normalize -> WMMA vs all 256 prototypes ->
// softmax/argmax in registers -> outputs + segment-sum atomics (x re-read
// from LDS, not HBM).
__global__ void __launch_bounds__(128)
assign_kernel(const float* __restrict__ x,
              const __bf16* __restrict__ pbf,
              float* __restrict__ soft,
              float* __restrict__ hard,
              float* __restrict__ sums,
              float* __restrict__ counts) {
    extern __shared__ __align__(16) unsigned char smem[];
    float*  x_raw = (float*)smem;
    __bf16* xbf   = (__bf16*)(smem + 65536);
    float*  inv_s = (float*)(smem + 98304);
    int*    lab_s = (int*)(smem + 98560);

    const int tid  = threadIdx.x;
    const int w    = tid >> 5;          // wave in block (0..3)
    const int ln   = tid & 31;          // lane
    const int half = ln >> 4;           // 0: lanes 0-15, 1: lanes 16-31
    const int li   = ln & 15;
    const long row_blk0 = (long)blockIdx.x * 64;   // first row of this block
    const long row0     = row_blk0 + (long)w * 16; // first row of this wave

    // Warm L2/WGP$ with the prototype table (global_prefetch_b8).
    __builtin_prefetch(pbf + (size_t)tid * 512, 0, 1);

    // ---- Phase 0: bulk async DMA of the 256x64 f32 x tile into LDS ----
#if __has_builtin(__builtin_amdgcn_tensor_load_to_lds)
    if (tid < 32) {
        const unsigned long long gaddr =
            (unsigned long long)(const void*)(x + row_blk0 * C_);
        const unsigned lds_base = (unsigned)(unsigned long long)(void*)x_raw;
        u32x4 g0;
        g0[0] = 1u;                                   // count=1, gather off
        g0[1] = lds_base;                             // lds_addr
        g0[2] = (unsigned)gaddr;                      // global_addr[31:0]
        g0[3] = (unsigned)((gaddr >> 32) & 0x1FFFFFFull) | (2u << 30); // addr[56:32] | type=2
        i32x8 g1 = {};
        g1[0] = (int)(2u << 16);          // data_size = 2 (4 bytes)
        g1[1] = (int)(256u << 16);        // tensor_dim0 = 256 (bits 79:48)
        g1[2] = (int)(64u << 16);         // tensor_dim1 = 64  (bits 111:80)
        g1[3] = (int)(256u << 16);        // tile_dim0 = 256   (bits 127:112)
        g1[4] = 64;                       // tile_dim1 = 64, tile_dim2 = 0
        g1[5] = 256;                      // tensor_dim0_stride = 256
        i32x4 z4 = {};
#if __has_include(<hip/amd_detail/amd_gfx1250_TDM.h>)
        i32x8 z8 = {};
        __builtin_amdgcn_tensor_load_to_lds(g0, g1, z4, z4, z8, 0);
#else
        __builtin_amdgcn_tensor_load_to_lds(g0, g1, z4, z4, 0);
#endif
        __builtin_amdgcn_s_wait_tensorcnt(0);
    }
#else
    for (int i = tid; i < 64 * C_ / 4; i += 128)
        ((float4*)x_raw)[i] = ((const float4*)(x + row_blk0 * C_))[i];
#endif
    __syncthreads();

    // ---- Phase 1: L2-normalize 16 rows (from LDS), stage bf16 x_norm ----
    for (int r = 0; r < 16; ++r) {
        const float* xr = x_raw + (w * 16 + r) * C_;
        const float4 a = reinterpret_cast<const float4*>(xr)[ln * 2];
        const float4 b = reinterpret_cast<const float4*>(xr)[ln * 2 + 1];
        float ss = a.x*a.x + a.y*a.y + a.z*a.z + a.w*a.w
                 + b.x*b.x + b.y*b.y + b.z*b.z + b.w*b.w;
#pragma unroll
        for (int off = 16; off >= 1; off >>= 1) ss += __shfl_xor(ss, off, 32);
        const float inv = 1.0f / fmaxf(sqrtf(ss), 1e-12f);
        __bf16* dst = xbf + (w * 16 + r) * C_ + ln * 8;
        dst[0] = (__bf16)(a.x * inv); dst[1] = (__bf16)(a.y * inv);
        dst[2] = (__bf16)(a.z * inv); dst[3] = (__bf16)(a.w * inv);
        dst[4] = (__bf16)(b.x * inv); dst[5] = (__bf16)(b.y * inv);
        dst[6] = (__bf16)(b.z * inv); dst[7] = (__bf16)(b.w * inv);
        if (ln == 0) inv_s[w * 16 + r] = inv;
    }
    __syncthreads();

    // ---- Phase 2: 16x256 score tile via v_wmma_f32_16x16x32_bf16 ----
    v8f acc[16];
#pragma unroll
    for (int t = 0; t < 16; ++t) { v8f z = {}; acc[t] = z; }

    const __bf16* xrow = xbf + (size_t)(w * 16 + li) * C_;
    const __bf16* pbase = pbf + (size_t)li * C_ + half * 16;
#pragma unroll
    for (int cc = 0; cc < 8; ++cc) {
        // A fragment (16x32 bf16): lanes 0-15 row=li K{0..15},
        //                          lanes 16-31 row=li K{16..31}
        Frag a;
        const int ao = cc * 32 + (half ? 16 : 0);
        a.u[0] = *reinterpret_cast<const v4u*>(xrow + ao);
        a.u[1] = *reinterpret_cast<const v4u*>(xrow + ao + 8);
        // B fragments (32x16 bf16): lane N=li, K 0..15 / 16..31 per half.
        // Software double-buffer so WMMA overlaps the next global fetch.
        Frag b0, b1;
        {
            const __bf16* pr = pbase + cc * 32;
            b0.u[0] = *reinterpret_cast<const v4u*>(pr);
            b0.u[1] = *reinterpret_cast<const v4u*>(pr + 8);
        }
#pragma unroll
        for (int kt = 0; kt < 16; kt += 2) {
            if (kt + 1 < 16) {
                const __bf16* pr = pbase + (size_t)(kt + 1) * 16 * C_ + cc * 32;
                b1.u[0] = *reinterpret_cast<const v4u*>(pr);
                b1.u[1] = *reinterpret_cast<const v4u*>(pr + 8);
            }
            acc[kt] = __builtin_amdgcn_wmma_f32_16x16x32_bf16(
                false, a.v, false, b0.v, (short)0, acc[kt], false, false);
            if (kt + 2 < 16) {
                const __bf16* pr = pbase + (size_t)(kt + 2) * 16 * C_ + cc * 32;
                b0.u[0] = *reinterpret_cast<const v4u*>(pr);
                b0.u[1] = *reinterpret_cast<const v4u*>(pr + 8);
            }
            acc[kt + 1] = __builtin_amdgcn_wmma_f32_16x16x32_bf16(
                false, a.v, false, b1.v, (short)0, acc[kt + 1], false, false);
        }
    }

    // ---- Phase 3: softmax(ALPHA*cos) + argmax per row (registers only) ----
    // D layout: acc[t][v] = score(row = v + 8*half, col = 16*t + li)
#pragma unroll
    for (int v = 0; v < 8; ++v) {
        const int  r    = v + half * 8;
        const long grow = row0 + r;

        float mv = -3.4e38f;
#pragma unroll
        for (int t = 0; t < 16; ++t) mv = fmaxf(mv, acc[t][v]);
#pragma unroll
        for (int off = 8; off >= 1; off >>= 1)
            mv = fmaxf(mv, __shfl_xor(mv, off, 32));   // stays within 16-lane half

        float bv = -3.4e38f; int bi = 0;
#pragma unroll
        for (int t = 0; t < 16; ++t) {
            const float s = acc[t][v];
            if (s > bv) { bv = s; bi = t * 16 + li; }   // first/smallest col on ties
        }
#pragma unroll
        for (int off = 8; off >= 1; off >>= 1) {
            const float ov = __shfl_xor(bv, off, 32);
            const int   oi = __shfl_xor(bi, off, 32);
            if (ov > bv || (ov == bv && oi < bi)) { bv = ov; bi = oi; }
        }

        float se = 0.f;
#pragma unroll
        for (int t = 0; t < 16; ++t) se += __expf(ALPHA * (acc[t][v] - mv));
#pragma unroll
        for (int off = 8; off >= 1; off >>= 1) se += __shfl_xor(se, off, 32);
        const float rinv = 1.0f / se;

        float* so = soft + grow * K_;
#pragma unroll
        for (int t = 0; t < 16; ++t)
            so[t * 16 + li] = __expf(ALPHA * (acc[t][v] - mv)) * rinv;

        if (li == 0) {
            hard[grow]        = (float)bi;
            lab_s[w * 16 + r] = bi;
        }
    }
    __syncthreads();

    // ---- Phase 4: segment-sum scatter of f32 x_norm (x from LDS) ----
    for (int r = 0; r < 16; ++r) {
        const int   lbl = lab_s[w * 16 + r];
        const float inv = inv_s[w * 16 + r];
        const float* xr = x_raw + (w * 16 + r) * C_;
        const float4 a = reinterpret_cast<const float4*>(xr)[ln * 2];
        const float4 b = reinterpret_cast<const float4*>(xr)[ln * 2 + 1];
        float* sb = sums + (size_t)lbl * C_ + ln * 8;
        atomicAdd(sb + 0, a.x * inv); atomicAdd(sb + 1, a.y * inv);
        atomicAdd(sb + 2, a.z * inv); atomicAdd(sb + 3, a.w * inv);
        atomicAdd(sb + 4, b.x * inv); atomicAdd(sb + 5, b.y * inv);
        atomicAdd(sb + 6, b.z * inv); atomicAdd(sb + 7, b.w * inv);
        if (ln == 0) atomicAdd(counts + lbl, 1.0f);
    }
}

// Kernel 3: momentum prototype update.
__global__ void __launch_bounds__(256)
finalize(const float* __restrict__ protos,
         const float* __restrict__ sums,
         const float* __restrict__ counts,
         float* __restrict__ out) {
    const int idx = blockIdx.x * 256 + threadIdx.x;
    const int k   = idx >> 8;
    const float cnt  = counts[k];
    const float p    = protos[idx];
    const float mean = sums[idx] / fmaxf(cnt, 1.0f);
    out[idx] = (cnt > 0.f) ? (MOM * p + (1.0f - MOM) * mean) : p;
}

extern "C" void kernel_launch(void* const* d_in, const int* in_sizes, int n_in,
                              void* d_out, int out_size, void* d_ws, size_t ws_size,
                              hipStream_t stream) {
    const float* x      = (const float*)d_in[0];   // (B,N,C) f32
    const float* protos = (const float*)d_in[1];   // (K,C) f32

    float* soft = (float*)d_out;                   // BN*K
    float* hard = soft + (size_t)BN_ * K_;         // BN
    float* pout = hard + BN_;                      // K*C

    char*   ws     = (char*)d_ws;
    __bf16* pbf    = (__bf16*)ws;
    float*  sums   = (float*)(ws + 131072);
    float*  counts = (float*)(ws + 131072 + (size_t)K_ * C_ * 4);

    // Allow >64KB dynamic LDS (host-side attribute set; not a stream op).
    (void)hipFuncSetAttribute((const void*)assign_kernel,
                              hipFuncAttributeMaxDynamicSharedMemorySize,
                              SMEM_BYTES);

    proto_prep<<<K_, C_, 0, stream>>>(protos, pbf, sums, counts);
    assign_kernel<<<BN_ / 64, 128, SMEM_BYTES, stream>>>(x, pbf, soft, hard,
                                                         sums, counts);
    finalize<<<K_, C_, 0, stream>>>(protos, sums, counts, pout);
}
